// DiffusionTransformer3DCLS_LearnPos_40132174413922
// MI455X (gfx1250) — matrix operand order repaired and checked by reading, not compiled, not run on hardware
//
#include <hip/hip_runtime.h>
#include <math.h>

#define Bv 8
#define Lv 5160
#define Cv 2
#define Dv 256
#define Hv 8
#define DEPTHv 8
#define FFv 1024
#define TEv 256
#define Mv (Bv*Lv)   // 41280 tokens total, divisible by 64

typedef __bf16 bf16;
typedef __attribute__((ext_vector_type(16))) __bf16 v16bf;
typedef __attribute__((ext_vector_type(8)))  __bf16 v8bf;
typedef __attribute__((ext_vector_type(8)))  float  v8f;

__device__ __forceinline__ float siluf(float x) { return x / (1.f + __expf(-x)); }
__device__ __forceinline__ float geluf(float x) { return 0.5f * x * (1.f + erff(x * 0.70710678118f)); }

// ---------------------------------------------------------------------------
// f32 -> bf16 conversion
// ---------------------------------------------------------------------------
__global__ void f2bf_kernel(bf16* __restrict__ dst, const float* __restrict__ src, int n) {
    int i = blockIdx.x * 256 + threadIdx.x;
    if (i < n) dst[i] = (bf16)src[i];
}

// ---------------------------------------------------------------------------
// WMMA GEMM:  out(M,N) = epilogue( A(M,K) @ W(N,K)^T + bias )
// mode 0: store bf16(out)            (K/V projections)
// mode 1: store bf16(gelu(out))      (MLP up-projection)
// mode 2: outF(M,N) += gate[b,n] * out   with b = m / Ltok   (MLP down + gate)
// Wave tile: 32(M) x 64(N). Per k-step: issue ALL 12 b128 loads (2 A frags +
// 4 B frags) before the 8 WMMAs so the compiler can retire loads with partial
// s_wait_loadcnt values and overlap VMEM latency under XDL issue.
// Block = 8 waves (2 M x 4 N) -> 64(M) x 256(N) block tile.
// ---------------------------------------------------------------------------
__global__ __launch_bounds__(256)
void gemm_wmma(const bf16* __restrict__ A, const bf16* __restrict__ W,
               const float* __restrict__ bias,
               bf16* __restrict__ outB, float* __restrict__ outF,
               const float* __restrict__ gate, int gstride,
               int M, int N, int K, int mode, int Ltok)
{
    int wave = threadIdx.x >> 5;
    int lane = threadIdx.x & 31;
    int lh   = lane & 15;
    int hi   = lane >> 4;

    int m0 = blockIdx.x * 64  + (wave >> 2) * 32;
    int n0 = blockIdx.y * 256 + (wave & 3)  * 64;

    v8f acc0[4] = {v8f{}, v8f{}, v8f{}, v8f{}};
    v8f acc1[4] = {v8f{}, v8f{}, v8f{}, v8f{}};

    const bf16* arow0 = A + (size_t)(m0 + lh) * K;        // rows m0 .. m0+15
    const bf16* arow1 = A + (size_t)(m0 + 16 + lh) * K;   // rows m0+16 .. m0+31
    const bf16* wbase = W + (size_t)(n0 + lh) * K;        // col n0+lh, +16K per subtile

    for (int kk = 0; kk < K; kk += 32) {
        // ---- issue all loads for this k-step first ----
        int kbA = kk + hi * 8;    // A: lanes 0-15 -> K base kk, lanes 16-31 -> kk+8
        int kbB = kk + hi * 8;    // B: lanes 0-15 -> K base kk, lanes 16-31 -> kk+8 (same as A)

        union { v8bf h[2]; v16bf v; } ua0, ua1, ub[4];
        ua0.h[0] = *(const v8bf*)(arow0 + kbA);
        ua0.h[1] = *(const v8bf*)(arow0 + kbA + 16);
        ua1.h[0] = *(const v8bf*)(arow1 + kbA);
        ua1.h[1] = *(const v8bf*)(arow1 + kbA + 16);
#pragma unroll
        for (int j = 0; j < 4; ++j) {
            const bf16* wrow = wbase + (size_t)(j * 16) * K + kbB;
            ub[j].h[0] = *(const v8bf*)(wrow);
            ub[j].h[1] = *(const v8bf*)(wrow + 16);
        }

        // prefetch next k-chunk of A into cache (global_prefetch_b8)
        if (kk + 32 < K) {
            __builtin_prefetch(arow0 + kbA + 32, 0, 3);
            __builtin_prefetch(arow1 + kbA + 32, 0, 3);
        }

        // ---- 8 WMMAs, B fragment reused across both M halves ----
#pragma unroll
        for (int j = 0; j < 4; ++j) {
            acc0[j] = __builtin_amdgcn_wmma_f32_16x16x32_bf16(
                false, ua0.v, false, ub[j].v, (short)0, acc0[j], false, false);
            acc1[j] = __builtin_amdgcn_wmma_f32_16x16x32_bf16(
                false, ua1.v, false, ub[j].v, (short)0, acc1[j], false, false);
        }
    }

#pragma unroll
    for (int j = 0; j < 4; ++j) {
        int n = n0 + j * 16 + lh;
        float bsv = bias ? bias[n] : 0.f;
#pragma unroll
        for (int i = 0; i < 8; ++i) {
            int m = m0 + i + hi * 8;
            float v = acc0[j][i] + bsv;
            if (mode == 2) {
                int b = m / Ltok;
                outF[(size_t)m * N + n] += gate[b * gstride + n] * v;
            } else {
                if (mode == 1) v = geluf(v);
                outB[(size_t)m * N + n] = (bf16)v;
            }
        }
#pragma unroll
        for (int i = 0; i < 8; ++i) {
            int m = m0 + 16 + i + hi * 8;
            float v = acc1[j][i] + bsv;
            if (mode == 2) {
                int b = m / Ltok;
                outF[(size_t)m * N + n] += gate[b * gstride + n] * v;
            } else {
                if (mode == 1) v = geluf(v);
                outB[(size_t)m * N + n] = (bf16)v;
            }
        }
    }
}

// ---------------------------------------------------------------------------
// Small f32 matmul: Y(Brows,N) = act_out( f(X)(Brows,K) @ W(N,K)^T + bias )
// silu_in applies SiLU to X elements; act_out: 0 none, 1 silu, 2 gelu
// ---------------------------------------------------------------------------
__global__ void small_mm(const float* __restrict__ X, const float* __restrict__ W,
                         const float* __restrict__ bias, float* __restrict__ Y,
                         int Brows, int N, int K, int act_out, int silu_in)
{
    int idx = blockIdx.x * 256 + threadIdx.x;
    if (idx >= Brows * N) return;
    int b = idx / N, n = idx % N;
    const float* x = X + (size_t)b * K;
    const float* w = W + (size_t)n * K;
    float acc = bias ? bias[n] : 0.f;
    for (int k = 0; k < K; ++k) {
        float xv = x[k];
        if (silu_in) xv = siluf(xv);
        acc += xv * w[k];
    }
    if (act_out == 1) acc = siluf(acc);
    else if (act_out == 2) acc = geluf(acc);
    Y[idx] = acc;
}

// ---------------------------------------------------------------------------
// Time embedding: sin/cos features
// ---------------------------------------------------------------------------
__global__ void time_sincos(const int* __restrict__ t, float* __restrict__ temb0)
{
    int i = blockIdx.x * 256 + threadIdx.x;
    if (i >= Bv * 128) return;
    int b = i >> 7, j = i & 127;
    float fr  = __expf(-logf(10000.f) * (float)j / 128.f);
    float ang = (float)t[b] * fr;
    temb0[b * TEv + j]       = sinf(ang);
    temb0[b * TEv + 128 + j] = cosf(ang);
}

// ---------------------------------------------------------------------------
// tokens = x_t @ in_w^T + in_b + pos
// ---------------------------------------------------------------------------
__global__ void token_embed(const float* __restrict__ x_t, const float* __restrict__ in_w,
                            const float* __restrict__ in_b, const float* __restrict__ pos,
                            float* __restrict__ tokens)
{
    int i = blockIdx.x * 256 + threadIdx.x;
    if (i >= Mv * Dv) return;
    int d = i % Dv;
    int ml = i / Dv;        // b*L + l
    int l = ml % Lv;
    tokens[i] = x_t[ml * 2] * in_w[d * 2] + x_t[ml * 2 + 1] * in_w[d * 2 + 1]
              + in_b[d] + pos[(size_t)l * Dv + d];
}

__global__ void bcast_cls(const float* __restrict__ src, float* __restrict__ dst, int n, int ncols)
{
    int i = blockIdx.x * 256 + threadIdx.x;
    if (i < n) dst[i] = src[i % ncols];
}

__global__ void vadd(const float* __restrict__ a, const float* __restrict__ b,
                     float* __restrict__ c, int n)
{
    int i = blockIdx.x * 256 + threadIdx.x;
    if (i < n) c[i] = a[i] + b[i];
}

// dst += gate[b, goff+c] * src   (gate stride gstride)
__global__ void gate_add(float* __restrict__ dst, const float* __restrict__ gate,
                         int gstride, int goff, const float* __restrict__ src,
                         int n, int ncols)
{
    int i = blockIdx.x * 256 + threadIdx.x;
    if (i >= n) return;
    int b = i / ncols, c = i % ncols;
    dst[i] += gate[b * gstride + goff + c] * src[i];
}

// ---------------------------------------------------------------------------
// cls AdaLN: hc = (1+s)*LN(cls)*lg+lb + sh  ; one block per batch row
// ---------------------------------------------------------------------------
__global__ __launch_bounds__(256)
void cls_ln_mod(const float* __restrict__ cls, const float* __restrict__ mod,
                const float* __restrict__ lg, const float* __restrict__ lb,
                float* __restrict__ hc)
{
    int b = blockIdx.x, d = threadIdx.x;
    __shared__ float r1[256], r2[256];
    float x = cls[b * Dv + d];
    r1[d] = x; r2[d] = x * x;
    __syncthreads();
    for (int s = 128; s; s >>= 1) {
        if (d < s) { r1[d] += r1[d + s]; r2[d] += r2[d + s]; }
        __syncthreads();
    }
    float mean = r1[0] * (1.f / Dv);
    float var  = r2[0] * (1.f / Dv) - mean * mean;
    float inv  = rsqrtf(var + 1e-5f);
    float ln = (x - mean) * inv * lg[d] + lb[d];
    hc[b * Dv + d] = (1.f + mod[b * 768 + d]) * ln + mod[b * 768 + 256 + d];
}

// ---------------------------------------------------------------------------
// token AdaLN -> bf16 h ; one wave (32 lanes) per token, 8 channels per lane
// ---------------------------------------------------------------------------
__global__ __launch_bounds__(256)
void token_modln(const float* __restrict__ tokens, const float* __restrict__ mod,
                 const float* __restrict__ lg, const float* __restrict__ lb,
                 bf16* __restrict__ h)
{
    int m = blockIdx.x * 8 + (threadIdx.x >> 5);
    int lane = threadIdx.x & 31;
    if (m >= Mv) return;
    int b = m / Lv;
    const float* x = tokens + (size_t)m * Dv;
    float vals[8], s = 0.f, s2 = 0.f;
#pragma unroll
    for (int j = 0; j < 8; ++j) {
        float v = x[j * 32 + lane];
        vals[j] = v; s += v; s2 += v * v;
    }
#pragma unroll
    for (int off = 16; off; off >>= 1) {
        s  += __shfl_xor(s,  off, 32);
        s2 += __shfl_xor(s2, off, 32);
    }
    float mean = s * (1.f / Dv);
    float var  = s2 * (1.f / Dv) - mean * mean;
    float inv  = rsqrtf(var + 1e-5f);
#pragma unroll
    for (int j = 0; j < 8; ++j) {
        int d = j * 32 + lane;
        float ln = (vals[j] - mean) * inv * lg[d] + lb[d];
        float hv = (1.f + mod[b * 768 + d]) * ln + mod[b * 768 + 256 + d];
        h[(size_t)m * Dv + d] = (bf16)hv;
    }
}

// ---------------------------------------------------------------------------
// CLS attention: one block per (b,h); q(32) vs K/V(L,32); softmax over L
// ---------------------------------------------------------------------------
__global__ __launch_bounds__(256)
void attn_cls(const float* __restrict__ q, const bf16* __restrict__ kbuf,
              const bf16* __restrict__ vbuf, float* __restrict__ o)
{
    int b = blockIdx.x >> 3, h = blockIdx.x & 7;
    int tid = threadIdx.x;
    __shared__ float sc[Lv];
    __shared__ float red[256];
    __shared__ float vacc[256 * 32];
    __shared__ float qs[32];
    if (tid < 32) qs[tid] = q[b * Dv + h * 32 + tid];
    __syncthreads();

    const float scale = 0.17677669529663687f; // 1/sqrt(32)
    float lmax = -1e30f;
    for (int l = tid; l < Lv; l += 256) {
        const bf16* kr = kbuf + ((size_t)(b * Lv + l)) * Dv + h * 32;
        float s = 0.f;
#pragma unroll
        for (int d = 0; d < 32; ++d) s += qs[d] * (float)kr[d];
        s *= scale;
        sc[l] = s;
        lmax = fmaxf(lmax, s);
    }
    red[tid] = lmax; __syncthreads();
    for (int st = 128; st; st >>= 1) {
        if (tid < st) red[tid] = fmaxf(red[tid], red[tid + st]);
        __syncthreads();
    }
    float gmax = red[0];
    __syncthreads();

    float acc[32];
#pragma unroll
    for (int d = 0; d < 32; ++d) acc[d] = 0.f;
    float wsum = 0.f;
    for (int l = tid; l < Lv; l += 256) {
        float w = __expf(sc[l] - gmax);
        wsum += w;
        const bf16* vr = vbuf + ((size_t)(b * Lv + l)) * Dv + h * 32;
#pragma unroll
        for (int d = 0; d < 32; ++d) acc[d] += w * (float)vr[d];
    }
    red[tid] = wsum; __syncthreads();
    for (int st = 128; st; st >>= 1) {
        if (tid < st) red[tid] += red[tid + st];
        __syncthreads();
    }
#pragma unroll
    for (int d = 0; d < 32; ++d) vacc[tid * 32 + d] = acc[d];
    __syncthreads();
    for (int st = 128; st; st >>= 1) {
        if (tid < st)
#pragma unroll
            for (int d = 0; d < 32; ++d) vacc[tid * 32 + d] += vacc[(tid + st) * 32 + d];
        __syncthreads();
    }
    if (tid < 32) o[b * Dv + h * 32 + tid] = vacc[tid] / red[0];
}

// ---------------------------------------------------------------------------
// Final: out = LN(tokens)*fg+fb @ out_w^T + out_b  ; one wave per token
// ---------------------------------------------------------------------------
__global__ __launch_bounds__(256)
void final_out(const float* __restrict__ tokens, const float* __restrict__ fg,
               const float* __restrict__ fb, const float* __restrict__ ow,
               const float* __restrict__ ob, float* __restrict__ out)
{
    int m = blockIdx.x * 8 + (threadIdx.x >> 5);
    int lane = threadIdx.x & 31;
    if (m >= Mv) return;
    const float* x = tokens + (size_t)m * Dv;
    float vals[8], s = 0.f, s2 = 0.f;
#pragma unroll
    for (int j = 0; j < 8; ++j) {
        float v = x[j * 32 + lane];
        vals[j] = v; s += v; s2 += v * v;
    }
#pragma unroll
    for (int off = 16; off; off >>= 1) {
        s  += __shfl_xor(s,  off, 32);
        s2 += __shfl_xor(s2, off, 32);
    }
    float mean = s * (1.f / Dv);
    float var  = s2 * (1.f / Dv) - mean * mean;
    float inv  = rsqrtf(var + 1e-5f);
    float p0 = 0.f, p1 = 0.f;
#pragma unroll
    for (int j = 0; j < 8; ++j) {
        int d = j * 32 + lane;
        float ln = (vals[j] - mean) * inv * fg[d] + fb[d];
        p0 += ln * ow[d];
        p1 += ln * ow[Dv + d];
    }
#pragma unroll
    for (int off = 16; off; off >>= 1) {
        p0 += __shfl_xor(p0, off, 32);
        p1 += __shfl_xor(p1, off, 32);
    }
    if (lane == 0) {
        out[(size_t)m * 2]     = p0 + ob[0];
        out[(size_t)m * 2 + 1] = p1 + ob[1];
    }
}

// ---------------------------------------------------------------------------
extern "C" void kernel_launch(void* const* d_in, const int* in_sizes, int n_in,
                              void* d_out, int out_size, void* d_ws, size_t ws_size,
                              hipStream_t stream)
{
    const float* x_t        = (const float*)d_in[0];
    const int*   t          = (const int*)  d_in[1];
    const float* in_w       = (const float*)d_in[2];
    const float* in_b       = (const float*)d_in[3];
    const float* pos        = (const float*)d_in[4];
    const float* cls_tok    = (const float*)d_in[5];
    const float* tp1_w      = (const float*)d_in[6];
    const float* tp1_b      = (const float*)d_in[7];
    const float* tp2_w      = (const float*)d_in[8];
    const float* tp2_b      = (const float*)d_in[9];
    const float* attn_in_w  = (const float*)d_in[10];
    const float* attn_in_b  = (const float*)d_in[11];
    const float* attn_out_w = (const float*)d_in[12];
    const float* attn_out_b = (const float*)d_in[13];
    const float* adaln_mod_w= (const float*)d_in[14];
    const float* adaln_mod_b= (const float*)d_in[15];
    const float* adaln_ln_g = (const float*)d_in[16];
    const float* adaln_ln_b = (const float*)d_in[17];
    const float* mlp_w1     = (const float*)d_in[18];
    const float* mlp_b1     = (const float*)d_in[19];
    const float* mlp_w2     = (const float*)d_in[20];
    const float* mlp_b2     = (const float*)d_in[21];
    const float* fin_g      = (const float*)d_in[22];
    const float* fin_b      = (const float*)d_in[23];
    const float* out_w      = (const float*)d_in[24];
    const float* out_b      = (const float*)d_in[25];
    float* out = (float*)d_out;

    // --- workspace carve-up ---
    char* wp = (char*)d_ws;
    auto alloc = [&](size_t bytes) -> void* {
        void* p = (void*)wp;
        wp += (bytes + 255) & ~(size_t)255;
        return p;
    };
    float* tokens = (float*)alloc((size_t)Mv * Dv * 4);
    bf16*  tbf    = (bf16*) alloc((size_t)Mv * Dv * 2);
    bf16*  hbuf   = (bf16*) alloc((size_t)Mv * Dv * 2);
    bf16*  kbuf   = (bf16*) alloc((size_t)Mv * Dv * 2);
    bf16*  vbuf   = (bf16*) alloc((size_t)Mv * Dv * 2);
    bf16*  y1     = (bf16*) alloc((size_t)Mv * FFv * 2);
    bf16*  wk_bf  = (bf16*) alloc((size_t)DEPTHv * Dv * Dv * 2);
    bf16*  wv_bf  = (bf16*) alloc((size_t)DEPTHv * Dv * Dv * 2);
    bf16*  w1_bf  = (bf16*) alloc((size_t)DEPTHv * FFv * Dv * 2);
    bf16*  w2_bf  = (bf16*) alloc((size_t)DEPTHv * Dv * FFv * 2);
    float* temb0  = (float*)alloc(Bv * TEv * 4);
    float* t1     = (float*)alloc(Bv * Dv * 4);
    float* temb   = (float*)alloc(Bv * Dv * 4);
    float* cls    = (float*)alloc(Bv * Dv * 4);
    float* hc     = (float*)alloc(Bv * Dv * 4);
    float* qb     = (float*)alloc(Bv * Dv * 4);
    float* obuf   = (float*)alloc(Bv * Dv * 4);
    float* proj   = (float*)alloc(Bv * Dv * 4);
    float* ymid   = (float*)alloc(Bv * FFv * 4);
    float* y2     = (float*)alloc(Bv * Dv * 4);
    float* cond   = (float*)alloc(Bv * Dv * 4);
    float* mod0   = (float*)alloc(Bv * 768 * 4);
    float* mod1   = (float*)alloc(Bv * 768 * 4);
    float* mod2   = (float*)alloc(Bv * 768 * 4);

    // --- weight conversion to bf16 (once per launch) ---
    for (int i = 0; i < DEPTHv; ++i) {
        f2bf_kernel<<<(Dv * Dv + 255) / 256, 256, 0, stream>>>(
            wk_bf + (size_t)i * Dv * Dv, attn_in_w + ((size_t)i * 768 + 256) * Dv, Dv * Dv);
        f2bf_kernel<<<(Dv * Dv + 255) / 256, 256, 0, stream>>>(
            wv_bf + (size_t)i * Dv * Dv, attn_in_w + ((size_t)i * 768 + 512) * Dv, Dv * Dv);
        f2bf_kernel<<<(FFv * Dv + 255) / 256, 256, 0, stream>>>(
            w1_bf + (size_t)i * FFv * Dv, mlp_w1 + (size_t)(i * 2 + 1) * FFv * Dv, FFv * Dv);
        f2bf_kernel<<<(Dv * FFv + 255) / 256, 256, 0, stream>>>(
            w2_bf + (size_t)i * Dv * FFv, mlp_w2 + (size_t)(i * 2 + 1) * Dv * FFv, Dv * FFv);
    }

    // --- time embedding ---
    time_sincos<<<(Bv * 128 + 255) / 256, 256, 0, stream>>>(t, temb0);
    small_mm<<<(Bv * Dv + 255) / 256, 256, 0, stream>>>(temb0, tp1_w, tp1_b, t1, Bv, Dv, TEv, 1, 0);
    small_mm<<<(Bv * Dv + 255) / 256, 256, 0, stream>>>(t1, tp2_w, tp2_b, temb, Bv, Dv, Dv, 0, 0);

    // --- token embedding + cls init ---
    token_embed<<<((size_t)Mv * Dv + 255) / 256, 256, 0, stream>>>(x_t, in_w, in_b, pos, tokens);
    bcast_cls<<<(Bv * Dv + 255) / 256, 256, 0, stream>>>(cls_tok, cls, Bv * Dv, Dv);

    dim3 gN256(Mv / 64, Dv / 256);   // (645, 1)
    dim3 gN1024(Mv / 64, FFv / 256); // (645, 4)

    for (int i = 0; i < DEPTHv; ++i) {
        const float* mw0 = adaln_mod_w + (size_t)(i * 3 + 0) * 768 * Dv;
        const float* mw1 = adaln_mod_w + (size_t)(i * 3 + 1) * 768 * Dv;
        const float* mw2 = adaln_mod_w + (size_t)(i * 3 + 2) * 768 * Dv;
        const float* mb0 = adaln_mod_b + (i * 3 + 0) * 768;
        const float* mb1 = adaln_mod_b + (i * 3 + 1) * 768;
        const float* mb2 = adaln_mod_b + (i * 3 + 2) * 768;
        const float* lg0 = adaln_ln_g + (i * 3 + 0) * Dv;
        const float* lg1 = adaln_ln_g + (i * 3 + 1) * Dv;
        const float* lg2 = adaln_ln_g + (i * 3 + 2) * Dv;
        const float* lb0 = adaln_ln_b + (i * 3 + 0) * Dv;
        const float* lb1 = adaln_ln_b + (i * 3 + 1) * Dv;
        const float* lb2 = adaln_ln_b + (i * 3 + 2) * Dv;

        // tokens -> bf16 for K/V projections
        f2bf_kernel<<<((size_t)Mv * Dv + 255) / 256, 256, 0, stream>>>(tbf, tokens, Mv * Dv);

        // K/V projections (WMMA)
        gemm_wmma<<<gN256, 256, 0, stream>>>(tbf, wk_bf + (size_t)i * Dv * Dv,
            attn_in_b + i * 768 + 256, kbuf, nullptr, nullptr, 0, Mv, Dv, Dv, 0, Lv);
        gemm_wmma<<<gN256, 256, 0, stream>>>(tbf, wv_bf + (size_t)i * Dv * Dv,
            attn_in_b + i * 768 + 512, vbuf, nullptr, nullptr, 0, Mv, Dv, Dv, 0, Lv);

        // cls attention branch
        small_mm<<<(Bv * 768 + 255) / 256, 256, 0, stream>>>(temb, mw0, mb0, mod0, Bv, 768, Dv, 0, 1);
        cls_ln_mod<<<Bv, 256, 0, stream>>>(cls, mod0, lg0, lb0, hc);
        small_mm<<<(Bv * Dv + 255) / 256, 256, 0, stream>>>(hc,
            attn_in_w + (size_t)i * 768 * Dv, attn_in_b + i * 768, qb, Bv, Dv, Dv, 0, 0);
        attn_cls<<<Bv * Hv, 256, 0, stream>>>(qb, kbuf, vbuf, obuf);
        small_mm<<<(Bv * Dv + 255) / 256, 256, 0, stream>>>(obuf,
            attn_out_w + (size_t)i * Dv * Dv, attn_out_b + i * Dv, proj, Bv, Dv, Dv, 0, 0);
        gate_add<<<(Bv * Dv + 255) / 256, 256, 0, stream>>>(cls, mod0, 768, 512, proj, Bv * Dv, Dv);

        // cls MLP branch
        small_mm<<<(Bv * 768 + 255) / 256, 256, 0, stream>>>(temb, mw1, mb1, mod1, Bv, 768, Dv, 0, 1);
        cls_ln_mod<<<Bv, 256, 0, stream>>>(cls, mod1, lg1, lb1, hc);
        small_mm<<<(Bv * FFv + 255) / 256, 256, 0, stream>>>(hc,
            mlp_w1 + (size_t)(i * 2) * FFv * Dv, mlp_b1 + (i * 2) * FFv, ymid, Bv, FFv, Dv, 2, 0);
        small_mm<<<(Bv * Dv + 255) / 256, 256, 0, stream>>>(ymid,
            mlp_w2 + (size_t)(i * 2) * Dv * FFv, mlp_b2 + (i * 2) * Dv, y2, Bv, Dv, FFv, 0, 0);
        gate_add<<<(Bv * Dv + 255) / 256, 256, 0, stream>>>(cls, mod1, 768, 512, y2, Bv * Dv, Dv);

        // token branch: cond = temb + cls ; mod2 = silu(cond) @ mw2^T + mb2
        vadd<<<(Bv * Dv + 255) / 256, 256, 0, stream>>>(temb, cls, cond, Bv * Dv);
        small_mm<<<(Bv * 768 + 255) / 256, 256, 0, stream>>>(cond, mw2, mb2, mod2, Bv, 768, Dv, 0, 1);
        token_modln<<<Mv / 8, 256, 0, stream>>>(tokens, mod2, lg2, lb2, hbuf);

        // token MLP (WMMA): up (GELU) then gated-down accumulate into tokens
        gemm_wmma<<<gN1024, 256, 0, stream>>>(hbuf, w1_bf + (size_t)i * FFv * Dv,
            mlp_b1 + (size_t)(i * 2 + 1) * FFv, y1, nullptr, nullptr, 0, Mv, FFv, Dv, 1, Lv);
        gemm_wmma<<<gN256, 256, 0, stream>>>(y1, w2_bf + (size_t)i * Dv * FFv,
            mlp_b2 + (size_t)(i * 2 + 1) * Dv, nullptr, tokens, mod2 + 512, 768, Mv, Dv, FFv, 2, Lv);
    }

    // --- final LN + output projection ---
    final_out<<<Mv / 8, 256, 0, stream>>>(tokens, fin_g, fin_b, out_w, out_b, out);
}
